// ParallelSMLSTMCell_83434034692712
// MI455X (gfx1250) — compile-verified, matrix operand order checked by reading, not run on hardware
//
#include <hip/hip_runtime.h>
#include <math.h>

typedef _Float16 h16;
typedef __attribute__((ext_vector_type(16))) _Float16 v16h;
typedef __attribute__((ext_vector_type(8)))  float    v8f;

// ---------------- WMMA helpers (gfx1250, wave32, 16x16x32 f16->f32) ----------------
__device__ __forceinline__ v8f wmma_f16(v16h a, v16h b, v8f c) {
  return __builtin_amdgcn_wmma_f32_16x16x32_f16(false, a, false, b, (short)0, c, false, false);
}
__device__ __forceinline__ v8f zero8() {
  v8f c;
#pragma unroll
  for (int i = 0; i < 8; ++i) c[i] = 0.f;
  return c;
}
// A row-major [16 x K] f16 (also B supplied as [N x K] row-major, i.e. B^T storage)
__device__ __forceinline__ v16h ld_rm_h(const h16* p, int ld, int lane) {
  int r = lane & 15, ko = (lane >> 4) << 3;
  v16h a;
#pragma unroll
  for (int i = 0; i < 8; ++i) { a[i] = p[r*ld + ko + i]; a[i+8] = p[r*ld + 16 + ko + i]; }
  return a;
}
__device__ __forceinline__ v16h ld_rm_f(const float* p, int ld, int lane) {
  int r = lane & 15, ko = (lane >> 4) << 3;
  v16h a;
#pragma unroll
  for (int i = 0; i < 8; ++i) { a[i] = (h16)p[r*ld + ko + i]; a[i+8] = (h16)p[r*ld + 16 + ko + i]; }
  return a;
}
// B stored [K x N] row-major
__device__ __forceinline__ v16h ld_kn_h(const h16* p, int ld, int lane) {
  int n = lane & 15, ko = (lane >> 4) << 3;
  v16h a;
#pragma unroll
  for (int i = 0; i < 8; ++i) { a[i] = p[(ko+i)*ld + n]; a[i+8] = p[(16+ko+i)*ld + n]; }
  return a;
}
__device__ __forceinline__ v16h ld_kn_f(const float* p, int ld, int lane) {
  int n = lane & 15, ko = (lane >> 4) << 3;
  v16h a;
#pragma unroll
  for (int i = 0; i < 8; ++i) { a[i] = (h16)p[(ko+i)*ld + n]; a[i+8] = (h16)p[(16+ko+i)*ld + n]; }
  return a;
}
// A[m][k] = p[k*ld + (moff+m)] * ks[k]  (transpose-read with per-k scale)
__device__ __forceinline__ v16h ld_cm_scale(const h16* p, int ld, const float* ks, int moff, int lane) {
  int m = moff + (lane & 15), ko = (lane >> 4) << 3;
  v16h a;
#pragma unroll
  for (int i = 0; i < 8; ++i) {
    int k0 = ko + i, k1 = 16 + ko + i;
    a[i]   = (h16)((float)p[k0*ld + m] * ks[k0]);
    a[i+8] = (h16)((float)p[k1*ld + m] * ks[k1]);
  }
  return a;
}
// causal masked attention operand: A[s][t] = (t<=s) ? S[s][t]*ig[t] : 0  (S is 32x32 f32)
__device__ __forceinline__ v16h ld_attnA(const float* Sm, const float* ig, int moff, int lane) {
  int m = moff + (lane & 15), ko = (lane >> 4) << 3;
  v16h a;
#pragma unroll
  for (int i = 0; i < 8; ++i) {
    int k0 = ko + i, k1 = 16 + ko + i;
    a[i]   = (h16)((k0 <= m) ? Sm[m*32 + k0] * ig[k0] : 0.f);
    a[i+8] = (h16)((k1 <= m) ? Sm[m*32 + k1] * ig[k1] : 0.f);
  }
  return a;
}
// A[s][t] = (t<=s) ? w[t] : 0
__device__ __forceinline__ v16h ld_rdsA(const float* w, int moff, int lane) {
  int m = moff + (lane & 15), ko = (lane >> 4) << 3;
  v16h a;
#pragma unroll
  for (int i = 0; i < 8; ++i) {
    int k0 = ko + i, k1 = 16 + ko + i;
    a[i]   = (h16)((k0 <= m) ? w[k0] : 0.f);
    a[i+8] = (h16)((k1 <= m) ? w[k1] : 0.f);
  }
  return a;
}
__device__ __forceinline__ v8f ld_c32(const float* p, int ld, int lane) {
  int n = lane & 15, mo = (lane >> 4) << 3;
  v8f c;
#pragma unroll
  for (int r = 0; r < 8; ++r) c[r] = p[(r+mo)*ld + n];
  return c;
}
__device__ __forceinline__ void st_c32(float* p, int ld, int lane, v8f c) {
  int n = lane & 15, mo = (lane >> 4) << 3;
#pragma unroll
  for (int r = 0; r < 8; ++r) p[(r+mo)*ld + n] = c[r];
}
__device__ __forceinline__ void st_c16(h16* p, int ld, int lane, v8f c) {
  int n = lane & 15, mo = (lane >> 4) << 3;
#pragma unroll
  for (int r = 0; r < 8; ++r) p[(r+mo)*ld + n] = (h16)c[r];
}

struct SMParams {
  const float *x, *Cprev, *nprev, *mprev, *lng, *lnb;
  const float *wq,*bq,*wk,*bk,*wvv,*bv,*wi,*bi,*wf,*bf,*wo,*bo;
  const float *rdw1,*rdb1,*rdg,*rdbe,*rdw2,*rdb2;
  const float *loglam,*logb,*cvec,*logd,*logstep,*volgate,*alphap,*slng,*slnb,*lmg,*lmb;
};

__global__ __launch_bounds__(256) void smlstm_kernel(SMParams P, float* __restrict__ out) {
  constexpr int S = 512, F = 32, T = 32, NC = 16, WIN = 102;
  constexpr float SQD = 8.0f;
  const int b    = blockIdx.x;
  const int tid  = threadIdx.x;
  const int lane = tid & 31;
  const int wid  = tid >> 5;

  __shared__ __align__(16) unsigned char lds[61440];
  float* SV   = (float*)(lds + 0);       // [512] volatility scalar per step
  float* Mst  = (float*)(lds + 2048);    // [64][64] f32 running Cacc state
  h16*   Qh   = (h16*)(lds + 18432);     // [32][64]
  h16*   Knh  = (h16*)(lds + 22528);     // [32][64]
  h16*   Vnh  = (h16*)(lds + 26624);     // [32][64]
  h16*   Oh   = (h16*)(lds + 30720);     // [32][64]
  h16*   LAh  = (h16*)(lds + 34816);     // [32][64] intra-chunk linear attention
  h16*   XNh  = (h16*)(lds + 34816);     // alias: xn [32][32] (dead before LA)
  h16*   RDSh = (h16*)(lds + 38912);     // [32][64] row-drive intra prefix
  h16*   MQh  = (h16*)(lds + 43008);     // [64][32] M@q + fcum*Cprev@q
  float* BUFA = (float*)(lds + 47104);   // [2048] f32 scratch
  h16*   YSh  = (h16*)(lds + 47104);     // alias: yscal [32][64] (after GEMMs)
  float* VEC  = (float*)(lds + 55296);   // small vectors

  float* vIG  = VEC +   0; float* vIGN = VEC +  32; float* vFG  = VEC +  64; float* vFC  = VEC +  96;
  float* vKF  = VEC + 128; float* vKS  = VEC + 160; float* vMF  = VEC + 192; float* vQS  = VEC + 224;
  float* vE   = VEC + 256; float* vE2  = VEC + 288; float* vNQI = VEC + 320; float* vNQO = VEC + 352;
  float* vNPQ = VEC + 384; float* vDEN = VEC + 416; float* vAS  = VEC + 448; float* vRM  = VEC + 480;
  float* vCT2 = VEC + 512; float* vGR  = VEC + 544; float* vGD  = VEC + 576; float* vFRO = VEC + 608;
  float* vCPQ = VEC + 640; float* vWT  = VEC + 672;
  float* vRDC = VEC + 704; float* vNST = VEC + 768; float* vCPR = VEC + 832; float* vAD  = VEC + 896;
  float* vBD  = VEC + 960; float* vVG  = VEC + 1024; float* vC  = VEC + 1088; float* vSLG = VEC + 1152;
  float* vSLB = VEC + 1216; float* vSC = VEC + 1280;  // scalars/carries

  // ---------- Phase 0: init, constants, rolling volatility ----------
  for (int i = tid; i < 512;  i += 256) SV[i]  = 0.f;
  for (int i = tid; i < 4096; i += 256) Mst[i] = 0.f;
  if (tid < 32) vSC[tid] = 0.f;
  __syncthreads();
  if (tid < 64) {
    int j = tid;
    float rs = 0.f, sq = 0.f;
    for (int k = 0; k < 64; ++k) { float cv = P.Cprev[b*4096 + j*64 + k]; rs += cv; sq += cv*cv; }
    vCPR[j] = rs; atomicAdd(&vSC[3], sq);
    float lam = -expf(P.loglam[j]);
    float st  = expf(P.logstep[0]);
    float ad  = (2.f + st*lam) / (2.f - st*lam);
    vAD[j] = ad;
    vBD[j] = st * (1.f + ad) * expf(P.logb[j]) * 0.5f;
    vVG[j] = 1.f / (1.f + expf(-P.volgate[j]));
    vC[j]  = P.cvec[j];
    vSLG[j] = P.slng[j]; vSLB[j] = P.slnb[j];
    vRDC[j] = 0.f; vNST[j] = 0.f;
  }
  if (tid == 64) {
    float a  = 1.f / (1.f + expf(-P.alphap[0]));
    float ds = expf(P.logd[0]);
    vSC[5] = a; vSC[6] = ds; vSC[7] = a + (1.f - a) * ds;   // gcoef
    vSC[0] = 1.f; vSC[1] = 0.f; vSC[2] = 0.f;               // fcum / cp / frob carries
    float mp = P.mprev[b]; vSC[4] = fminf(fmaxf(mp, -100.f), 100.f);
  }
  __syncthreads();
  if (tid < 32) {  // rolling-window unbiased std (window=102), summed over features
    int f = tid; float a = 0.f, a2 = 0.f;
    for (int s = 0; s < S; ++s) {
      float xv = P.x[(b*S + s)*F + f];
      a += xv; a2 += xv*xv;
      if (s >= WIN) { float xo = P.x[(b*S + s - WIN)*F + f]; a -= xo; a2 -= xo*xo; }
      if (s >= WIN - 1) {
        float var = (a2 - a*a*(1.f/WIN)) * (1.f/(WIN-1));
        atomicAdd(&SV[s], sqrtf(fmaxf(var, 0.f)));
      }
    }
  }
  __syncthreads();
  float sv0 = SV[WIN-1];
  __syncthreads();
  for (int i = tid; i < WIN-1; i += 256) SV[i] = sv0;
  __syncthreads();
  { float lt = 0.f; for (int i = tid; i < S; i += 256) lt += SV[i]; atomicAdd(&vSC[8], lt); }
  __syncthreads();
  { float denm = vSC[8]*(1.f/(512.f*32.f)) + 1e-6f;
    float scl = (1.f/32.f)/denm;
    for (int i = tid; i < S; i += 256) SV[i] *= scl; }
  __syncthreads();

  float hreg[16];     // SSM hidden state: row=tid>>2, hidden block=(tid&3)*16
#pragma unroll
  for (int i = 0; i < 16; ++i) hreg[i] = 0.f;

  auto gemm_proj = [&](const float* Wg) {   // [32 x 32] @ [32 x 64] (weights [N x K] row-major)
    int mt = wid >> 2, nt = wid & 3;
    v16h a = ld_rm_h(XNh + mt*16*32, 32, lane);
    v16h bb = ld_rm_f(Wg + nt*16*32, 32, lane);
    v8f c = zero8();
    c = wmma_f16(a, bb, c);
    st_c32(BUFA + mt*16*64 + nt*16, 64, lane, c);
  };

  // ================= chunk loop =================
  for (int ch = 0; ch < NC; ++ch) {
    const int s0 = ch * T;

    // --- xn: LayerNorm over F=32 ---
    {
      int s = tid >> 3, f0 = (tid & 7) * 4;
      float xv[4]; float sm = 0.f, sq = 0.f;
#pragma unroll
      for (int ff = 0; ff < 4; ++ff) { xv[ff] = P.x[(b*S + s0 + s)*F + f0 + ff]; sm += xv[ff]; sq += xv[ff]*xv[ff]; }
      sm += __shfl_xor(sm,1,32); sm += __shfl_xor(sm,2,32); sm += __shfl_xor(sm,4,32);
      sq += __shfl_xor(sq,1,32); sq += __shfl_xor(sq,2,32); sq += __shfl_xor(sq,4,32);
      float mu = sm*(1.f/32.f);
      float rstd = rsqrtf(sq*(1.f/32.f) - mu*mu + 1e-5f);
#pragma unroll
      for (int ff = 0; ff < 4; ++ff) {
        int f = f0 + ff;
        XNh[s*32 + f] = (h16)((xv[ff]-mu)*rstd*P.lng[f] + P.lnb[f]);
      }
    }
    __syncthreads();

    // --- gates it/ft -> ig, fg ---
    if (tid < 32) {
      int s = tid; float it = P.bi[0], ft = P.bf[0];
      for (int f = 0; f < 32; ++f) { float xv = (float)XNh[s*32+f]; it += xv*P.wi[f]; ft += xv*P.wf[f]; }
      float mp = vSC[4];
      float mt = fmaxf(ft, mp + ft);
      float ci = fminf(fmaxf(it - mt, -15.f), 15.f);
      float cf = fminf(fmaxf(ft + mp - mt, -15.f), 15.f);
      float ig = expf(ci);
      vIGN[s] = ig; vIG[s] = fminf(ig, 1.f);
      vFG[s] = expf(cf);
    }
    __syncthreads();
    if (tid == 0) { float fc = vSC[0]; for (int s = 0; s < T; ++s) { fc *= vFG[s]; vFC[s] = fc; } }
    __syncthreads();

    // --- projections (WMMA) + per-s postprocessing ---
    gemm_proj(P.wq);
    __syncthreads();
    {
      int s = tid >> 3, d0 = (tid & 7) * 8; float ps = 0.f;
#pragma unroll
      for (int dd = 0; dd < 8; ++dd) { int d = d0+dd; float qv = BUFA[s*64+d] + P.bq[d]; Qh[s*64+d] = (h16)qv; ps += qv; }
      ps += __shfl_xor(ps,1,32); ps += __shfl_xor(ps,2,32); ps += __shfl_xor(ps,4,32);
      if ((tid & 7) == 0) vQS[s] = ps;
    }
    __syncthreads();
    gemm_proj(P.wk);
    __syncthreads();
    {
      int s = tid >> 3, d0 = (tid & 7) * 8;
      float kv[8]; float ss = 0.f;
#pragma unroll
      for (int dd = 0; dd < 8; ++dd) { int d = d0+dd; kv[dd] = (BUFA[s*64+d] + P.bk[d]) * 0.125f; ss += kv[dd]*kv[dd]; }
      ss += __shfl_xor(ss,1,32); ss += __shfl_xor(ss,2,32); ss += __shfl_xor(ss,4,32);
      float nrm = sqrtf(ss);
      float inv = SQD / (nrm + 1e-12f);
      float pk = 0.f;
#pragma unroll
      for (int dd = 0; dd < 8; ++dd) { int d = d0+dd; float kn = kv[dd]*inv; Knh[s*64+d] = (h16)kn; pk += kn; }
      pk += __shfl_xor(pk,1,32); pk += __shfl_xor(pk,2,32); pk += __shfl_xor(pk,4,32);
      if ((tid & 7) == 0) { vKS[s] = pk; vKF[s] = nrm * 0.125f; }
    }
    __syncthreads();
    gemm_proj(P.wvv);
    __syncthreads();
    {
      int s = tid >> 3, d0 = (tid & 7) * 8;
      float vv[8]; float ss = 0.f;
#pragma unroll
      for (int dd = 0; dd < 8; ++dd) { int d = d0+dd; vv[dd] = BUFA[s*64+d] + P.bv[d]; ss += vv[dd]*vv[dd]; }
      ss += __shfl_xor(ss,1,32); ss += __shfl_xor(ss,2,32); ss += __shfl_xor(ss,4,32);
      float inv = SQD / (sqrtf(ss) + 1e-6f);
#pragma unroll
      for (int dd = 0; dd < 8; ++dd) { int d = d0+dd; Vnh[s*64+d] = (h16)(vv[dd]*inv); }
    }
    __syncthreads();
    gemm_proj(P.wo);
    __syncthreads();
    {
      int s = tid >> 3, d0 = (tid & 7) * 8;
#pragma unroll
      for (int dd = 0; dd < 8; ++dd) { int d = d0+dd; Oh[s*64+d] = (h16)(1.f/(1.f + expf(-(BUFA[s*64+d] + P.bo[d])))); }
    }
    __syncthreads();
    gemm_proj(P.rdw1);
    __syncthreads();
    if (tid < 32) {   // regime head: LN -> gelu -> 4 logits -> softmax -> mf
      int s = tid;
      float mu = 0.f;
      for (int j = 0; j < 64; ++j) mu += BUFA[s*64+j] + P.rdb1[j];
      mu *= (1.f/64.f);
      float v2 = 0.f;
      for (int j = 0; j < 64; ++j) { float d0 = BUFA[s*64+j] + P.rdb1[j] - mu; v2 += d0*d0; }
      float rstd = rsqrtf(v2*(1.f/64.f) + 1e-5f);
      float l0 = P.rdb2[0], l1 = P.rdb2[1], l2 = P.rdb2[2], l3 = P.rdb2[3];
      for (int j = 0; j < 64; ++j) {
        float zn = (BUFA[s*64+j] + P.rdb1[j] - mu)*rstd*P.rdg[j] + P.rdbe[j];
        float ge = 0.5f*zn*(1.f + erff(zn*0.70710678f));
        l0 += ge*P.rdw2[j]; l1 += ge*P.rdw2[64+j]; l2 += ge*P.rdw2[128+j]; l3 += ge*P.rdw2[192+j];
      }
      float mx = fmaxf(fmaxf(l0,l1), fmaxf(l2,l3));
      float e0 = expf(l0-mx), e1 = expf(l1-mx), e2 = expf(l2-mx), e3 = expf(l3-mx);
      vMF[s] = e0 / (e0+e1+e2+e3);
    }
    __syncthreads();

    // --- n_state·q and n_prev·q (before state update) ---
    if (tid < 32) {
      int s = tid; float nq = 0.f, npq = 0.f;
      for (int d = 0; d < 64; ++d) { float qv = (float)Qh[s*64+d]; nq += vNST[d]*qv; npq += P.nprev[b*64+d]*qv; }
      vNQO[s] = nq; vNPQ[s] = npq;
    }
    __syncthreads();

    // --- Vn@M -> cross term e_t = vn^T M kn ---
    {
      int mt = wid >> 2, nt = wid & 3;
      v16h a0 = ld_rm_h(Vnh + mt*16*64, 64, lane);
      v16h a1 = ld_rm_h(Vnh + mt*16*64 + 32, 64, lane);
      v16h b0 = ld_kn_f(Mst + nt*16, 64, lane);
      v16h b1 = ld_kn_f(Mst + 32*64 + nt*16, 64, lane);
      v8f c = zero8(); c = wmma_f16(a0,b0,c); c = wmma_f16(a1,b1,c);
      st_c32(BUFA + mt*16*64 + nt*16, 64, lane, c);
    }
    __syncthreads();
    if (tid < 32) { int t = tid; float e = 0.f;
      for (int j = 0; j < 64; ++j) e += BUFA[t*64+j]*(float)Knh[t*64+j];
      vE[t] = e; }
    __syncthreads();
    // --- Vn@Cprev -> e2_t = vn^T Cprev kn ---
    {
      int mt = wid >> 2, nt = wid & 3;
      const float* Cp = P.Cprev + b*4096;
      v16h a0 = ld_rm_h(Vnh + mt*16*64, 64, lane);
      v16h a1 = ld_rm_h(Vnh + mt*16*64 + 32, 64, lane);
      v16h b0 = ld_kn_f(Cp + nt*16, 64, lane);
      v16h b1 = ld_kn_f(Cp + 32*64 + nt*16, 64, lane);
      v8f c = zero8(); c = wmma_f16(a0,b0,c); c = wmma_f16(a1,b1,c);
      st_c32(BUFA + mt*16*64 + nt*16, 64, lane, c);
    }
    __syncthreads();
    if (tid < 32) { int t = tid; float e = 0.f;
      for (int j = 0; j < 64; ++j) e += BUFA[t*64+j]*(float)Knh[t*64+j];
      vE2[t] = e; }
    __syncthreads();

    // --- Gram matrices VV^T (waves 0-3) and KK^T (waves 4-7) ---
    {
      int w4 = wid & 3; int mt = w4 >> 1, nt = w4 & 1;
      const h16* src = (wid < 4) ? Vnh : Knh;
      float* dst = (wid < 4) ? BUFA : (BUFA + 1024);
      v16h a0 = ld_rm_h(src + mt*16*64, 64, lane);
      v16h a1 = ld_rm_h(src + mt*16*64 + 32, 64, lane);
      v16h b0 = ld_rm_h(src + nt*16*64, 64, lane);
      v16h b1 = ld_rm_h(src + nt*16*64 + 32, 64, lane);
      v8f c = zero8(); c = wmma_f16(a0,b0,c); c = wmma_f16(a1,b1,c);
      st_c32(dst + mt*16*32 + nt*16, 32, lane, c);
    }
    __syncthreads();
    if (tid < 32) {
      int t = tid; float g = 0.f;
      for (int t2 = 0; t2 < t; ++t2) g += vIG[t]*vIG[t2]*BUFA[t*32+t2]*BUFA[1024 + t*32 + t2];
      vGR[t] = g;
      vGD[t] = vIG[t]*vIG[t]*BUFA[t*32+t]*BUFA[1024 + t*32 + t];
    }
    __syncthreads();
    if (tid == 0) {  // exact Frobenius / <Cprev,Cacc> prefixes
      float run = 0.f, pe = 0.f, pe2 = 0.f;
      for (int s = 0; s < T; ++s) {
        pe  += vIG[s]*vE[s];
        pe2 += vIG[s]*vE2[s];
        run += 2.f*vGR[s] + vGD[s];
        vFRO[s] = vSC[2] + 2.f*pe + run;
        vCPQ[s] = vSC[1] + pe2;
      }
    }
    __syncthreads();

    // --- Sgem = Q·Kn^T [32x32] ---
    if (wid < 4) {
      int mt = wid >> 1, nt = wid & 1;
      v16h a0 = ld_rm_h(Qh + mt*16*64, 64, lane);
      v16h a1 = ld_rm_h(Qh + mt*16*64 + 32, 64, lane);
      v16h b0 = ld_rm_h(Knh + nt*16*64, 64, lane);
      v16h b1 = ld_rm_h(Knh + nt*16*64 + 32, 64, lane);
      v8f c = zero8(); c = wmma_f16(a0,b0,c); c = wmma_f16(a1,b1,c);
      st_c32(BUFA + mt*16*32 + nt*16, 32, lane, c);
    }
    __syncthreads();
    if (tid < 32) {
      int s = tid; float nq = 0.f;
      for (int t = 0; t <= s; ++t) nq += vIGN[t]*vKF[t]*BUFA[s*32+t];
      vNQI[s] = nq;
      vWT[s] = vIG[s]*vKS[s];
    }
    __syncthreads();

    // --- LA = (mask·ig·Sgem) @ Vn : intra-chunk linear attention ---
    {
      int mt = wid >> 2, nt = wid & 3;
      v16h a = ld_attnA(BUFA, vIG, mt*16, lane);
      v16h bb = ld_kn_h(Vnh + nt*16, 64, lane);
      v8f c = zero8(); c = wmma_f16(a, bb, c);
      st_c16(LAh + mt*16*64 + nt*16, 64, lane, c);
    }
    __syncthreads();
    // --- RDS = (mask·ig·knsum) @ Vn : row-drive prefix for SSM ---
    {
      int mt = wid >> 2, nt = wid & 3;
      v16h a = ld_rdsA(vWT, mt*16, lane);
      v16h bb = ld_kn_h(Vnh + nt*16, 64, lane);
      v8f c = zero8(); c = wmma_f16(a, bb, c);
      st_c16(RDSh + mt*16*64 + nt*16, 64, lane, c);
    }
    __syncthreads();

    // --- Cprev@Q^T, scale by fcum, accumulate M@Q^T -> MQh ---
    {
      int mt = wid >> 1, nt = wid & 1;
      const float* Cp = P.Cprev + b*4096;
      v16h a0 = ld_rm_f(Cp + mt*16*64, 64, lane);
      v16h a1 = ld_rm_f(Cp + mt*16*64 + 32, 64, lane);
      v16h b0 = ld_rm_h(Qh + nt*16*64, 64, lane);
      v16h b1 = ld_rm_h(Qh + nt*16*64 + 32, 64, lane);
      v8f c = zero8(); c = wmma_f16(a0,b0,c); c = wmma_f16(a1,b1,c);
      st_c32(BUFA + mt*16*32 + nt*16, 32, lane, c);
    }
    __syncthreads();
    {
      int i = tid >> 2, sl = (tid & 3) * 8;
#pragma unroll
      for (int ss = 0; ss < 8; ++ss) { int s = sl + ss; BUFA[i*32+s] *= vFC[s]; }
    }
    __syncthreads();
    {
      int mt = wid >> 1, nt = wid & 1;
      v16h a0 = ld_rm_f(Mst + mt*16*64, 64, lane);
      v16h a1 = ld_rm_f(Mst + mt*16*64 + 32, 64, lane);
      v16h b0 = ld_rm_h(Qh + nt*16*64, 64, lane);
      v16h b1 = ld_rm_h(Qh + nt*16*64 + 32, 64, lane);
      v8f c = ld_c32(BUFA + mt*16*32 + nt*16, 32, lane);
      c = wmma_f16(a0,b0,c); c = wmma_f16(a1,b1,c);
      st_c16(MQh + mt*16*32 + nt*16, 32, lane, c);
    }
    __syncthreads();

    // --- sequential SSM scan (row=tid>>2, 16 hidden per thread) ---
    {
      int row = tid >> 2, jb = (tid & 3) * 16;
      for (int s = 0; s < T; ++s) {
        float u = (vFC[s]*vCPR[row] + vRDC[row] + (float)RDSh[s*64+row]) * (1.f/64.f);
        float svv = SV[s0 + s];
        float sm = 0.f, sq = 0.f; float tmp[16];
#pragma unroll
        for (int jj = 0; jj < 16; ++jj) {
          int j = jb + jj;
          float hv = vAD[j]*hreg[jj] + vBD[j]*u;
          tmp[jj] = hv; sm += hv; sq += hv*hv;
        }
        sm += __shfl_xor(sm,1,32); sm += __shfl_xor(sm,2,32);
        sq += __shfl_xor(sq,1,32); sq += __shfl_xor(sq,2,32);
        float mu = sm*(1.f/64.f);
        float rstd = rsqrtf(sq*(1.f/64.f) - mu*mu + 1e-5f);
        float ys = 0.f;
#pragma unroll
        for (int jj = 0; jj < 16; ++jj) {
          int j = jb + jj;
          float hn = ((tmp[jj]-mu)*rstd*vSLG[j] + vSLB[j]) / (1.f + vVG[j]*svv);
          hreg[jj] = hn; ys += hn*vC[j];
        }
        ys += __shfl_xor(ys,1,32); ys += __shfl_xor(ys,2,32);
        if ((tid & 3) == 0) YSh[s*64+row] = (h16)ys;
      }
    }
    __syncthreads();

    // --- F1: per-s scalars ---
    if (tid < 32) {
      int s = tid; float fc = vFC[s];
      vDEN[s] = fmaxf(fabsf(fc*vNPQ[s] + vNQO[s] + vNQI[s]), 1.f);
      vCT2[s] = fc*fc*vSC[3] + 2.f*fc*vCPQ[s] + vFRO[s];
      float mf = vMF[s];
      vAS[s] = mf*vSC[7] + (1.f - mf);
      vRM[s] = mf*(1.f - vSC[5]);
    }
    __syncthreads();

    // --- F2: readout: norm, matvec, stabilizer, LN, gate ---
    {
      int s = tid >> 3, i0 = (tid & 7) * 8;
      float fc = vFC[s], rm = vRM[s], A = vAS[s];
      float rr[8]; float sA = 0.f, sB = 0.f;
#pragma unroll
      for (int ii = 0; ii < 8; ++ii) {
        int i = i0 + ii;
        float ri = rm * (float)YSh[s*64+i];
        float rows = fc*vCPR[i] + vRDC[i] + (float)RDSh[s*64+i];
        rr[ii] = ri; sA += ri*rows; sB += ri*ri;
      }
      sA += __shfl_xor(sA,1,32); sA += __shfl_xor(sA,2,32); sA += __shfl_xor(sA,4,32);
      sB += __shfl_xor(sB,1,32); sB += __shfl_xor(sB,2,32); sB += __shfl_xor(sB,4,32);
      float fin2 = A*A*vCT2[s] + 2.f*A*sA + 64.f*sB;
      float scl = SQD / (sqrtf(fmaxf(fin2, 0.f)) + 1e-6f);
      float invden = 1.f / vDEN[s];
      float qs = vQS[s];
      float hv[8]; float hs = 0.f, hq = 0.f;
#pragma unroll
      for (int ii = 0; ii < 8; ++ii) {
        int i = i0 + ii;
        float ctq = (float)MQh[i*32+s] + (float)LAh[s*64+i];
        float ht = scl*(A*ctq + rr[ii]*qs)*invden;
        hv[ii] = ht; hs += ht; hq += ht*ht;
      }
      hs += __shfl_xor(hs,1,32); hs += __shfl_xor(hs,2,32); hs += __shfl_xor(hs,4,32);
      hq += __shfl_xor(hq,1,32); hq += __shfl_xor(hq,2,32); hq += __shfl_xor(hq,4,32);
      float mu = hs*(1.f/64.f);
      float rstd = rsqrtf(hq*(1.f/64.f) - mu*mu + 1e-5f);
#pragma unroll
      for (int ii = 0; ii < 8; ++ii) {
        int i = i0 + ii;
        float ov = ((hv[ii]-mu)*rstd*P.lmg[i] + P.lmb[i]) * (float)Oh[s*64+i];
        out[(b*S + s0 + s)*64 + i] = ov;
      }
    }
    __syncthreads();

    // --- carries + state update M += (ig·Vn)^T Kn ---
    if (tid < 64) {
      int i = tid; float acc = 0.f;
      for (int t = 0; t < T; ++t) acc += vIG[t]*vKS[t]*(float)Vnh[t*64+i];
      vRDC[i] += acc;
    } else if (tid < 128) {
      int d = tid - 64; float acc = 0.f;
      for (int t = 0; t < T; ++t) acc += vIGN[t]*vKF[t]*(float)Knh[t*64+d];
      vNST[d] += acc;
    } else if (tid == 128) {
      vSC[0] = vFC[T-1]; vSC[1] = vCPQ[T-1]; vSC[2] = vFRO[T-1];
    }
    __syncthreads();
    {
#pragma unroll
      for (int t2 = 0; t2 < 2; ++t2) {
        int tile = wid*2 + t2, mt = tile >> 2, nt = tile & 3;
        v16h a = ld_cm_scale(Vnh, 64, vIG, mt*16, lane);
        v16h bb = ld_kn_h(Knh + nt*16, 64, lane);
        v8f c = ld_c32(Mst + mt*16*64 + nt*16, 64, lane);
        c = wmma_f16(a, bb, c);
        st_c32(Mst + mt*16*64 + nt*16, 64, lane, c);
      }
    }
    __syncthreads();
  }
}

extern "C" void kernel_launch(void* const* d_in, const int* in_sizes, int n_in,
                              void* d_out, int out_size, void* d_ws, size_t ws_size,
                              hipStream_t stream) {
  (void)in_sizes; (void)n_in; (void)d_ws; (void)ws_size; (void)out_size;
  SMParams P;
  P.x      = (const float*)d_in[0];  P.Cprev  = (const float*)d_in[1];
  P.nprev  = (const float*)d_in[2];  P.mprev  = (const float*)d_in[3];
  P.lng    = (const float*)d_in[4];  P.lnb    = (const float*)d_in[5];
  P.wq     = (const float*)d_in[6];  P.bq     = (const float*)d_in[7];
  P.wk     = (const float*)d_in[8];  P.bk     = (const float*)d_in[9];
  P.wvv    = (const float*)d_in[10]; P.bv     = (const float*)d_in[11];
  P.wi     = (const float*)d_in[12]; P.bi     = (const float*)d_in[13];
  P.wf     = (const float*)d_in[14]; P.bf     = (const float*)d_in[15];
  P.wo     = (const float*)d_in[16]; P.bo     = (const float*)d_in[17];
  P.rdw1   = (const float*)d_in[18]; P.rdb1   = (const float*)d_in[19];
  P.rdg    = (const float*)d_in[20]; P.rdbe   = (const float*)d_in[21];
  P.rdw2   = (const float*)d_in[22]; P.rdb2   = (const float*)d_in[23];
  P.loglam = (const float*)d_in[24]; P.logb   = (const float*)d_in[25];
  P.cvec   = (const float*)d_in[26]; P.logd   = (const float*)d_in[27];
  P.logstep= (const float*)d_in[28]; P.volgate= (const float*)d_in[29];
  P.alphap = (const float*)d_in[30]; P.slng   = (const float*)d_in[31];
  P.slnb   = (const float*)d_in[32]; P.lmg    = (const float*)d_in[33];
  P.lmb    = (const float*)d_in[34];
  smlstm_kernel<<<dim3(8), dim3(256), 0, stream>>>(P, (float*)d_out);
}
